// CCN_22686017257848
// MI455X (gfx1250) — matrix operand order repaired and checked by reading, not compiled, not run on hardware
//
#include <hip/hip_runtime.h>
#include <hip/hip_bf16.h>

// MI455X / gfx1250, wave32. Fused SNN:
//  Phase 1: one wave per (n,d) site keeps the A=256-wide SDC/LIF/synapse state
//           in registers across all T=32 steps; only s2 spikes hit memory.
//  Phase 2: one persistent 512-thread workgroup runs the MLP head with
//           v_wmma_f32_16x16x32_f16. Layer-1 weights live in registers
//           (128 VGPRs/wave), layer-2 weights stream from L2 each timestep
//           (anti-LICM via runtime-zero stride so they can't be hoisted+spilled).

typedef _Float16 v16h __attribute__((ext_vector_type(16)));
typedef _Float16 v8h  __attribute__((ext_vector_type(8)));
typedef float    v8f  __attribute__((ext_vector_type(8)));

#define T_STEPS 32
#define NB      16
#define DD      256
#define AA      256
#define F1D     512
#define F2D     256

// LDS row strides padded by 8 halfs (16B) to skew banks, keep 16B alignment.
#define S_IN_STRIDE  264
#define S_MID_STRIDE 520
#define S_OUT_STRIDE 264

// ---------------------------------------------------------------------------
// Pack W[K][Nf] (f32 row-major) into WMMA-B f16 layout:
// Bp[kt][nt][lane][h] = W[kt*32 + (lane<16?0:16) + h][nt*16 + (lane&15)]
// so phase-2 B-tile loads are contiguous 32B per lane.
// ---------------------------------------------------------------------------
__global__ __launch_bounds__(256)
void ccn_pack_b_f16(const float* __restrict__ W, _Float16* __restrict__ Bp,
                    int K, int Nf) {
  int idx = blockIdx.x * blockDim.x + threadIdx.x;
  int total = K * Nf;
  if (idx >= total) return;
  int h    = idx & 15;
  int lane = (idx >> 4) & 31;
  int rest = idx >> 9;            // kt*NT + nt
  int NT   = Nf >> 4;
  int nt   = rest % NT;
  int kt   = rest / NT;
  int k = kt * 32 + ((lane < 16) ? 0 : 16) + h;
  int f = nt * 16 + (lane & 15);
  Bp[idx] = (_Float16)W[k * Nf + f];
}

// ---------------------------------------------------------------------------
// Phase 1: SDC linear (2->A) + exp filter + LIF + synapse filter + pool + LIF.
// One wave32 per (n,d); lane j-strip holds channels a = j*32 + lane.
// ---------------------------------------------------------------------------
__global__ __launch_bounds__(256)
void ccn_phase1(const float* __restrict__ x,
                const float* __restrict__ W_sdc,
                const float* __restrict__ b_sdc,
                const float* __restrict__ w_syn,
                const float* __restrict__ W_pool,
                const float* __restrict__ b_pool,
                float* __restrict__ s2buf) {
  const int lane = threadIdx.x & 31;
  const int wave = threadIdx.x >> 5;
  const int p = blockIdx.x * (blockDim.x >> 5) + wave;   // 0 .. N*D-1
  const int n = p >> 8;
  const int d = p & 255;

  float w0[8], w1[8], bs[8], wp[8];
#pragma unroll
  for (int j = 0; j < 8; ++j) {
    int a = j * 32 + lane;
    w0[j] = W_sdc[a];          // W_sdc[0][a]
    w1[j] = W_sdc[AA + a];     // W_sdc[1][a]
    bs[j] = b_sdc[a];
    wp[j] = W_pool[a];         // W_pool[a][0]
  }
  // SynapseFilter decay = 1 - sigmoid(w_syn) = 1 / (1 + exp(w_syn))
  const float dsyn = 1.0f / (1.0f + __expf(w_syn[0]));
  const float bp = b_pool[0];

  float cur[8] = {0.f}, vm[8] = {0.f}, syn[8] = {0.f};
  float v2 = 0.f;

  for (int t = 0; t < T_STEPS; ++t) {
    const float* xp = x + (((t * NB + n) * 2 + 0) * DD + d);
    const float x0 = xp[0];
    const float x1 = xp[DD];
    float zp = 0.f;
#pragma unroll
    for (int j = 0; j < 8; ++j) {
      float h = fmaf(x0, w0[j], fmaf(x1, w1[j], bs[j]));
      cur[j] = fmaf(cur[j], 0.5f, h);            // i = i*(1-1/tau_sdc) + h
      vm[j]  = 0.5f * (vm[j] + cur[j]);          // v += (i - v)/tau_lif
      float s = (vm[j] >= 1.0f) ? 1.0f : 0.0f;   // atan_spike fwd = Heaviside
      vm[j] *= (1.0f - s);                       // hard reset
      syn[j] = fmaf(syn[j], dsyn, s);            // synapse filter
      zp = fmaf(syn[j], wp[j], zp);              // pool partial dot
    }
    // wave32 butterfly reduce -> zp replicated in all lanes
#pragma unroll
    for (int off = 16; off > 0; off >>= 1)
      zp += __shfl_xor(zp, off, 32);
    float z = zp + bp;
    v2 = 0.5f * (v2 + z);
    float s2 = (v2 >= 1.0f) ? 1.0f : 0.0f;
    v2 *= (1.0f - s2);
    if (lane == 0) s2buf[(t * NB + n) * DD + d] = s2;
  }
}

// ---------------------------------------------------------------------------
// Phase 2: persistent workgroup (16 waves). Per timestep:
//   s2(16x256) --WMMA--> LIF(16x512) --WMMA--> LIF(16x256) --dot--> cumsum.
// Membranes persist in registers in the WMMA C/D layout.
// t_stride0 is always 0 at runtime; it makes the Bp2 addresses formally
// t-dependent so the compiler cannot hoist (and then spill) the weight loads.
// ---------------------------------------------------------------------------
__global__ __launch_bounds__(512)
void ccn_phase2(const float* __restrict__ s2buf,
                const _Float16* __restrict__ Bp1,
                const _Float16* __restrict__ Bp2,
                const float* __restrict__ b_f1,
                const float* __restrict__ b_f2,
                const float* __restrict__ W_out,
                const float* __restrict__ b_out,
                float* __restrict__ out,
                int t_stride0) {
  __shared__ alignas(16) _Float16 sIn [NB * S_IN_STRIDE];   // s2 spikes  (f16)
  __shared__ alignas(16) _Float16 sMid[NB * S_MID_STRIDE];  // h1 spikes  (f16)
  __shared__ alignas(16) _Float16 sOut[NB * S_OUT_STRIDE];  // h2 spikes  (f16)

  const int tid  = threadIdx.x;
  const int lane = tid & 31;
  const int wave = tid >> 5;             // 0..15
  const int msel = (lane < 16) ? 0 : 8;  // C/D row-half select
  const int nloc = lane & 15;            // column within a 16-wide tile

  // ---- Preload layer-1 weights into registers: this wave's two column
  // tiles (ntile = wave and wave+16), all 8 k-tiles. 16 x v16h = 128 VGPRs.
  v16h wb1[16];
#pragma unroll
  for (int kt = 0; kt < 8; ++kt) {
    wb1[2 * kt + 0] = *(const v16h*)&Bp1[((kt * 32 + wave) * 32 + lane) * 16];
    wb1[2 * kt + 1] = *(const v16h*)&Bp1[((kt * 32 + wave + 16) * 32 + lane) * 16];
  }

  // Persistent LIF membranes (C/D layout: elem e -> row e+msel, col nloc).
  v8f vm1a = {0.f,0.f,0.f,0.f,0.f,0.f,0.f,0.f};  // layer1 ntile = wave
  v8f vm1b = {0.f,0.f,0.f,0.f,0.f,0.f,0.f,0.f};  // layer1 ntile = wave+16
  v8f vm2  = {0.f,0.f,0.f,0.f,0.f,0.f,0.f,0.f};  // layer2 ntile = wave
  float cum = 0.f;                               // NonSpikingIF readout (n=wave)

  const float bf1a = b_f1[wave * 16 + nloc];
  const float bf1b = b_f1[(wave + 16) * 16 + nloc];
  const float bf2  = b_f2[wave * 16 + nloc];
  float wout[8];
#pragma unroll
  for (int j = 0; j < 8; ++j) wout[j] = W_out[j * 32 + lane];
  const float bo = b_out[0];

#pragma clang loop unroll(disable)
  for (int t = 0; t < T_STEPS; ++t) {
    // Formally t-dependent weight base (t_stride0 == 0 at runtime).
    const _Float16* Bp2t = Bp2 + (size_t)t * (size_t)t_stride0;

    // ---- stage 0: s2[t] -> f16 LDS (uniform 8 iterations, no divergence) --
    for (int e = tid; e < NB * DD; e += 512) {
      int m = e >> 8, f = e & 255;
      sIn[m * S_IN_STRIDE + f] = (_Float16)s2buf[(t * NB + m) * DD + f];
    }
    __syncthreads();

    // ---- stage 1: C1 = S2(16x256) x W_f1(256x512), 2 column tiles / wave,
    //      B operands register-resident (wb1) -------------------------------
    v8f acc0 = {0.f,0.f,0.f,0.f,0.f,0.f,0.f,0.f};
    v8f acc1 = {0.f,0.f,0.f,0.f,0.f,0.f,0.f,0.f};
#pragma unroll
    for (int kt = 0; kt < 8; ++kt) {
      // A tile (16x32 f16): lanes<16 hold K {0..15}, lanes>=16 {16..31}
      const _Float16* ap = &sIn[nloc * S_IN_STRIDE + kt * 32 + 2 * msel];
      union { v16h v; v8h h[2]; } au;
      au.h[0] = *(const v8h*)(ap);        // 16B ds load
      au.h[1] = *(const v8h*)(ap + 8);    // 16B ds load
      acc0 = __builtin_amdgcn_wmma_f32_16x16x32_f16(false, au.v, false,
                                                    wb1[2 * kt + 0],
                                                    (short)0, acc0, false, false);
      acc1 = __builtin_amdgcn_wmma_f32_16x16x32_f16(false, au.v, false,
                                                    wb1[2 * kt + 1],
                                                    (short)0, acc1, false, false);
    }
    // LIF layer 1 -> spikes into sMid
#pragma unroll
    for (int e = 0; e < 8; ++e) {
      int m = e + msel;
      float p0 = acc0[e] + bf1a;
      vm1a[e] = 0.5f * (vm1a[e] + p0);
      float s0 = (vm1a[e] >= 1.0f) ? 1.0f : 0.0f;
      vm1a[e] *= (1.0f - s0);
      sMid[m * S_MID_STRIDE + wave * 16 + nloc] = (_Float16)s0;
      float p1 = acc1[e] + bf1b;
      vm1b[e] = 0.5f * (vm1b[e] + p1);
      float s1 = (vm1b[e] >= 1.0f) ? 1.0f : 0.0f;
      vm1b[e] *= (1.0f - s1);
      sMid[m * S_MID_STRIDE + (wave + 16) * 16 + nloc] = (_Float16)s1;
    }
    __syncthreads();

    // ---- stage 2: C2 = H1(16x512) x W_f2(512x256), 1 column tile / wave,
    //      B operands streamed from L2 (not hoistable -> no scratch spill) --
    v8f acc2 = {0.f,0.f,0.f,0.f,0.f,0.f,0.f,0.f};
#pragma unroll
    for (int kt = 0; kt < 16; ++kt) {
      const _Float16* ap = &sMid[nloc * S_MID_STRIDE + kt * 32 + 2 * msel];
      union { v16h v; v8h h[2]; } au;
      au.h[0] = *(const v8h*)(ap);
      au.h[1] = *(const v8h*)(ap + 8);
      const v16h b = *(const v16h*)&Bp2t[((kt * 16 + wave) * 32 + lane) * 16];
      acc2 = __builtin_amdgcn_wmma_f32_16x16x32_f16(false, au.v, false, b,
                                                    (short)0, acc2, false, false);
    }
    __syncthreads();
    // LIF layer 2 -> spikes into sOut
#pragma unroll
    for (int e = 0; e < 8; ++e) {
      int m = e + msel;
      float p = acc2[e] + bf2;
      vm2[e] = 0.5f * (vm2[e] + p);
      float s = (vm2[e] >= 1.0f) ? 1.0f : 0.0f;
      vm2[e] *= (1.0f - s);
      sOut[m * S_OUT_STRIDE + wave * 16 + nloc] = (_Float16)s;
    }
    __syncthreads();

    // ---- stage 3: o[t][n=wave] = h2 . W_out + b_out ; cumsum over t -------
    float acc = 0.f;
#pragma unroll
    for (int j = 0; j < 8; ++j) {
      int f = j * 32 + lane;
      acc = fmaf((float)sOut[wave * S_OUT_STRIDE + f], wout[j], acc);
    }
#pragma unroll
    for (int off = 16; off > 0; off >>= 1)
      acc += __shfl_xor(acc, off, 32);
    cum += acc + bo;                       // NonSpikingIF: v_t = v_{t-1} + o_t
    if (lane == 0) out[t * NB + wave] = cum;
    __syncthreads();                       // protect sOut/sIn for next t
  }
}

// ---------------------------------------------------------------------------
extern "C" void kernel_launch(void* const* d_in, const int* in_sizes, int n_in,
                              void* d_out, int out_size, void* d_ws, size_t ws_size,
                              hipStream_t stream) {
  (void)in_sizes; (void)n_in; (void)out_size; (void)ws_size;
  const float* x      = (const float*)d_in[0];   // [32,16,2,256]
  const float* W_sdc  = (const float*)d_in[1];   // [2,256]
  const float* b_sdc  = (const float*)d_in[2];   // [256]
  const float* w_syn  = (const float*)d_in[3];   // [1]
  const float* W_pool = (const float*)d_in[4];   // [256,1]
  const float* b_pool = (const float*)d_in[5];   // [1]
  const float* W_f1   = (const float*)d_in[6];   // [256,512]
  const float* b_f1   = (const float*)d_in[7];   // [512]
  const float* W_f2   = (const float*)d_in[8];   // [512,256]
  const float* b_f2   = (const float*)d_in[9];   // [256]
  const float* W_out  = (const float*)d_in[10];  // [256,1]
  const float* b_out  = (const float*)d_in[11];  // [1]
  float* out = (float*)d_out;                    // [32,16,1]

  char* ws = (char*)d_ws;
  float*    s2buf = (float*)ws;                       // 32*16*256 f32 = 512 KB
  _Float16* Bp1   = (_Float16*)(ws + 524288);         // 256*512 f16  = 256 KB
  _Float16* Bp2   = (_Float16*)(ws + 786432);         // 512*256 f16  = 256 KB

  // Weight packing (f32 -> WMMA-B f16), 131072 elements each.
  ccn_pack_b_f16<<<512, 256, 0, stream>>>(W_f1, Bp1, 256, F1D);
  ccn_pack_b_f16<<<512, 256, 0, stream>>>(W_f2, Bp2, F1D, F2D);

  // Phase 1: 4096 waves (one per (n,d)), 8 waves per 256-thread block.
  ccn_phase1<<<512, 256, 0, stream>>>(x, W_sdc, b_sdc, w_syn, W_pool, b_pool,
                                      s2buf);

  // Phase 2: single persistent workgroup, 16 waves, WMMA MLP head.
  // Last arg is the anti-LICM runtime zero (deterministic, always 0).
  ccn_phase2<<<1, 512, 0, stream>>>(s2buf, Bp1, Bp2, b_f1, b_f2, W_out, b_out,
                                    out, 0);
}
